// Transformer_89335319757430
// MI455X (gfx1250) — matrix operand order repaired and checked by reading, not compiled, not run on hardware
//
#include <hip/hip_runtime.h>
#include <hip/hip_bf16.h>
#include <math.h>

typedef __attribute__((ext_vector_type(16))) __bf16 v16bf;
typedef __attribute__((ext_vector_type(8)))  float  v8f;
typedef __attribute__((ext_vector_type(4)))  float  vf4;
typedef __attribute__((ext_vector_type(4)))  __bf16 v4bf;

#define L_   4
#define H_   16
#define D_   1024
#define HS_  64
#define DFF_ 5120
#define V_   32000
#define B_   2
#define T_   1024
#define EPS_ 1e-5f

enum { BM_F32 = 0, BM_HEADED = 1, BM_BF16T = 2, BM_BF16 = 3 };
enum { EP_F32 = 0, EP_QKV = 1, EP_O = 2, EP_PROJ = 3, EP_FFN1 = 4 };

union Frag  { v16bf v; uint4 q[2]; };
union Pack4 { v4bf h; uint2 u; };

// ---------------------------------------------------------------------------
// Templated bf16 WMMA GEMM: C[M,N] = A[M,K](bf16) * B[K,N] (+bias, epilogue).
// 256 threads (8 waves), tile 128x128, K-step 32, ping-pong double-buffered
// LDS so the HBM weight stream overlaps the WMMA pipe (one barrier/K-step).
// B tile is stored TRANSPOSED in LDS (Bt[col][k]) so both A and B fragments
// are 2x contiguous ds_load_b128 per 16x16x32 operand. All global staging
// pointers advance by constant strides (no per-step index math).
// ---------------------------------------------------------------------------
template <int BMODE, int EMODE>
__global__ __launch_bounds__(256)
void gemm_t(const __bf16* __restrict__ A, long long aStrideZ, int lda,
            const float* __restrict__ Bf, const __bf16* __restrict__ Bb,
            long long bStrideZ, int ldb,
            const float* __restrict__ bias, const float* __restrict__ res,
            float* __restrict__ outF, __bf16* __restrict__ outB,
            long long outStrideZ, int ldc,
            int M, int N, int K)
{
  __shared__ __align__(16) __bf16 As[2][128][40];  // A[row][k]   (80B stride)
  __shared__ __align__(16) __bf16 Bt[2][128][40];  // B^T[col][k] (80B stride)

  const int tid  = threadIdx.x;
  const int wave = tid >> 5;
  const int lane = tid & 31;
  const int hf   = lane >> 4;      // lane half
  const int ln16 = lane & 15;
  const int wm   = wave & 3;       // wave row -> rows wm*32..+32
  const int wn   = wave >> 2;      // wave col -> cols wn*64..+64
  const int m0   = blockIdx.y * 128;
  const int n0   = blockIdx.x * 128;
  const int z    = blockIdx.z;

  A += (long long)z * aStrideZ;
  if (BMODE == BM_F32 || BMODE == BM_HEADED) Bf += (long long)z * bStrideZ;
  else                                       Bb += (long long)z * bStrideZ;

  v8f acc[2][4];
  #pragma unroll
  for (int i = 0; i < 2; ++i)
    #pragma unroll
    for (int j = 0; j < 4; ++j)
      #pragma unroll
      for (int e = 0; e < 8; ++e) acc[i][j][e] = 0.0f;

  // A staging: 2 threads/row, 16 bf16 (32B) each
  const int ar = tid >> 1;
  const int ac = (tid & 1) * 16;

  // B staging: 4x4 micro-transpose; kk0 = (tid>>5)*4, c0 = (tid&31)*4.
  // Clamp the column group so every load is in-bounds (branch-free); columns
  // >= N produce garbage that the epilogue discards.
  const int kk0 = (tid >> 5) * 4;
  const int c0  = (tid & 31) * 4;
  const int cCl = (n0 + c0 < N) ? (n0 + c0) : (N - 4);

  // ---- incrementally advanced staging pointers (constant K-step strides) ----
  const __bf16* aPtr = A + (long long)(m0 + ar) * lda + ac;          // += 32
  const float*  bfPtr = nullptr;
  const __bf16* bbPtr = nullptr;
  long long bStep = 0;
  if constexpr (BMODE == BM_F32) {
    bfPtr = Bf + (long long)kk0 * ldb + cCl;             bStep = 32LL * ldb;
  } else if constexpr (BMODE == BM_HEADED) {
    bfPtr = Bf + ((long long)(cCl >> 6) * K + kk0) * 64 + (cCl & 63);
    bStep = 32LL * 64;
  } else if constexpr (BMODE == BM_BF16T) {
    bbPtr = Bb + (long long)cCl * ldb + kk0;             bStep = 32;
  } else { // BM_BF16
    bbPtr = Bb + (long long)kk0 * ldb + cCl;             bStep = 32LL * ldb;
  }

  // ---- staging register state (ping-pong pipeline) ----
  uint4 aR0, aR1;
  vf4   bF[4];   // BM_F32 / BM_HEADED raw fp32
  Pack4 bH[4];   // BM_BF16T / BM_BF16 raw bf16

  auto loadTiles = [&]() {
    aR0 = ((const uint4*)aPtr)[0];
    aR1 = ((const uint4*)aPtr)[1];
    aPtr += 32;
    if constexpr (BMODE == BM_F32 || BMODE == BM_HEADED) {
      const int rstride = (BMODE == BM_F32) ? ldb : 64;
      #pragma unroll
      for (int i = 0; i < 4; ++i)
        bF[i] = *(const vf4*)(bfPtr + (long long)i * rstride);
      __builtin_prefetch(bfPtr + bStep, 0, 1);   // next K-step of HBM stream
      bfPtr += bStep;
    } else if constexpr (BMODE == BM_BF16T) {
      // B[k][n] = Kmat[n*ldb + k] : column-contiguous in global
      #pragma unroll
      for (int cc = 0; cc < 4; ++cc)
        bH[cc].h = *(const v4bf*)(bbPtr + (long long)cc * ldb);
      bbPtr += bStep;
    } else { // BM_BF16 row-major
      #pragma unroll
      for (int i = 0; i < 4; ++i)
        bH[i].h = *(const v4bf*)(bbPtr + (long long)i * ldb);
      bbPtr += bStep;
    }
  };

  auto storeTiles = [&](int buf) {
    uint4* dst = (uint4*)&As[buf][ar][ac];
    dst[0] = aR0; dst[1] = aR1;
    if constexpr (BMODE == BM_F32 || BMODE == BM_HEADED) {
      #pragma unroll
      for (int cc = 0; cc < 4; ++cc) {
        Pack4 p;
        p.h[0] = (__bf16)bF[0][cc]; p.h[1] = (__bf16)bF[1][cc];
        p.h[2] = (__bf16)bF[2][cc]; p.h[3] = (__bf16)bF[3][cc];
        *(uint2*)&Bt[buf][c0 + cc][kk0] = p.u;
      }
    } else if constexpr (BMODE == BM_BF16T) {
      #pragma unroll
      for (int cc = 0; cc < 4; ++cc)
        *(uint2*)&Bt[buf][c0 + cc][kk0] = bH[cc].u;
    } else {
      #pragma unroll
      for (int cc = 0; cc < 4; ++cc) {
        Pack4 p;
        p.h[0] = bH[0].h[cc]; p.h[1] = bH[1].h[cc];
        p.h[2] = bH[2].h[cc]; p.h[3] = bH[3].h[cc];
        *(uint2*)&Bt[buf][c0 + cc][kk0] = p.u;
      }
    }
  };

  // ---- pipelined main loop: one barrier per K-step ----
  const int nsteps = K >> 5;
  loadTiles();
  int buf = 0;
  for (int step = 0; step < nsteps; ++step) {
    storeTiles(buf);
    __syncthreads();
    if (step + 1 < nsteps) loadTiles();   // overlaps the WMMA block below

    Frag fa[2];
    #pragma unroll
    for (int s = 0; s < 2; ++s) {
      const int row = wm * 32 + s * 16 + ln16;
      fa[s].q[0] = *(const uint4*)&As[buf][row][hf * 8];       // K = hf*8..+7
      fa[s].q[1] = *(const uint4*)&As[buf][row][16 + hf * 8];  // K = 16+hf*8..+7
    }
    Frag fb[4];
    #pragma unroll
    for (int cb = 0; cb < 4; ++cb) {
      const int col = wn * 64 + cb * 16 + ln16;
      fb[cb].q[0] = *(const uint4*)&Bt[buf][col][hf * 8];       // K = hf*8..+7
      fb[cb].q[1] = *(const uint4*)&Bt[buf][col][16 + hf * 8];  // K = 16+hf*8..+7
    }
    #pragma unroll
    for (int s = 0; s < 2; ++s)
      #pragma unroll
      for (int cb = 0; cb < 4; ++cb)
        acc[s][cb] = __builtin_amdgcn_wmma_f32_16x16x32_bf16(
            false, fa[s].v, false, fb[cb].v, (short)0, acc[s][cb], false, false);
    buf ^= 1;
  }

  // ---- epilogue (compile-time specialized) ----
  #pragma unroll
  for (int s = 0; s < 2; ++s) {
    #pragma unroll
    for (int cb = 0; cb < 4; ++cb) {
      const int c = n0 + wn * 64 + cb * 16 + ln16;
      if (c >= N) continue;
      #pragma unroll
      for (int r = 0; r < 8; ++r) {
        const int m = m0 + wm * 32 + s * 16 + hf * 8 + r;
        float v = acc[s][cb][r];
        if (bias) v += bias[c];
        if constexpr (EMODE == EP_F32) {
          outF[(long long)z * outStrideZ + (long long)m * ldc + c] = v;
        } else if constexpr (EMODE == EP_QKV) {   // [b,h,t,e]; m=b*T+t, c=h*64+e
          const int b = m >> 10, t = m & 1023, h = c >> 6, e = c & 63;
          outB[(((long long)(b * H_ + h)) * T_ + t) * HS_ + e] = (__bf16)v;
        } else if constexpr (EMODE == EP_O) {     // z=b*H+h: out[b, t, h*64+e]
          const int b = z >> 4, h = z & 15;
          outB[((long long)(b * T_ + m)) * D_ + h * HS_ + c] = (__bf16)v;
        } else if constexpr (EMODE == EP_PROJ) {  // x = residual + proj
          const long long idx = (long long)m * ldc + c;
          outF[idx] = v + res[idx];
        } else if constexpr (EMODE == EP_FFN1) {  // leaky_relu -> bf16
          const float r2 = v > 0.0f ? v : 0.01f * v;
          outB[(long long)m * ldc + c] = (__bf16)r2;
        }
      }
    }
  }
}

// ---------------------------------------------------------------------------
// Elementwise kernels
// ---------------------------------------------------------------------------
__device__ inline float block_reduce_sum(float v, float* red) {
  const int tid = threadIdx.x;
  red[tid] = v; __syncthreads();
  #pragma unroll
  for (int off = 128; off > 0; off >>= 1) {
    if (tid < off) red[tid] += red[tid + off];
    __syncthreads();
  }
  const float r = red[0];
  __syncthreads();
  return r;
}

__device__ inline float block_reduce_max(float v, float* red) {
  const int tid = threadIdx.x;
  red[tid] = v; __syncthreads();
  #pragma unroll
  for (int off = 128; off > 0; off >>= 1) {
    if (tid < off) red[tid] = fmaxf(red[tid], red[tid + off]);
    __syncthreads();
  }
  const float r = red[0];
  __syncthreads();
  return r;
}

__global__ void embed_kernel(const int* __restrict__ idx,
                             const float* __restrict__ tok,
                             const float* __restrict__ pos,
                             float* __restrict__ x) {
  const long long i = (long long)blockIdx.x * blockDim.x + threadIdx.x;
  if (i >= (long long)B_ * T_ * D_) return;
  const int d = (int)(i % D_);
  const long long bt = i / D_;
  const int t = (int)(bt % T_);
  x[i] = tok[(long long)idx[bt] * D_ + d] + pos[(long long)t * D_ + d];
}

// row LayerNorm (D=1024) -> bf16 output
__global__ __launch_bounds__(256)
void ln_bf16_kernel(const float* __restrict__ x, const float* __restrict__ g,
                    const float* __restrict__ b, __bf16* __restrict__ out) {
  __shared__ float red[256];
  const int row = blockIdx.x, tid = threadIdx.x;
  const float* xr = x + (long long)row * D_;
  float v[4], s = 0.0f;
  #pragma unroll
  for (int i = 0; i < 4; ++i) { v[i] = xr[tid + i * 256]; s += v[i]; }
  const float mu = block_reduce_sum(s, red) * (1.0f / D_);
  s = 0.0f;
  #pragma unroll
  for (int i = 0; i < 4; ++i) { const float d = v[i] - mu; s += d * d; }
  const float rstd = rsqrtf(block_reduce_sum(s, red) * (1.0f / D_) + EPS_);
  __bf16* orow = out + (long long)row * D_;
  #pragma unroll
  for (int i = 0; i < 4; ++i) {
    const int c = tid + i * 256;
    orow[c] = (__bf16)((v[i] - mu) * rstd * g[c] + b[c]);
  }
}

// x += LN2( LN1(y, g1,b1), g2,b2 )  — the quirky FFN tail of the reference
__global__ __launch_bounds__(256)
void ffn_post_kernel(const float* __restrict__ y,
                     const float* __restrict__ g1, const float* __restrict__ b1,
                     const float* __restrict__ g2, const float* __restrict__ b2,
                     float* __restrict__ x) {
  __shared__ float red[256];
  const int row = blockIdx.x, tid = threadIdx.x;
  const float* yr = y + (long long)row * D_;
  float v[4], s = 0.0f;
  #pragma unroll
  for (int i = 0; i < 4; ++i) { v[i] = yr[tid + i * 256]; s += v[i]; }
  const float mu = block_reduce_sum(s, red) * (1.0f / D_);
  s = 0.0f;
  #pragma unroll
  for (int i = 0; i < 4; ++i) { const float d = v[i] - mu; s += d * d; }
  const float rstd = rsqrtf(block_reduce_sum(s, red) * (1.0f / D_) + EPS_);
  float zv[4]; s = 0.0f;
  #pragma unroll
  for (int i = 0; i < 4; ++i) {
    const int c = tid + i * 256;
    zv[i] = (v[i] - mu) * rstd * g1[c] + b1[c];
    s += zv[i];
  }
  const float mu2 = block_reduce_sum(s, red) * (1.0f / D_);
  s = 0.0f;
  #pragma unroll
  for (int i = 0; i < 4; ++i) { const float d = zv[i] - mu2; s += d * d; }
  const float rstd2 = rsqrtf(block_reduce_sum(s, red) * (1.0f / D_) + EPS_);
  float* xr = x + (long long)row * D_;
  #pragma unroll
  for (int i = 0; i < 4; ++i) {
    const int c = tid + i * 256;
    xr[c] += (zv[i] - mu2) * rstd2 * g2[c] + b2[c];
  }
}

// scale + (causal mask) + softmax over rows of T=1024 fp32 scores;
// rewrites probabilities IN PLACE as bf16 with row stride 2048 elements
// (the bf16 row occupies the first half of its own fp32 row -> row-private).
__global__ __launch_bounds__(256)
void softmax_kernel(float* __restrict__ scores, int causal) {
  __shared__ float red[256];
  const long long row = blockIdx.x;           // b*H*T + h*T + t
  const int t = (int)(row % T_);
  const int tid = threadIdx.x;
  float* sr = scores + row * (long long)T_;
  float v[4];
  float mx = -1e30f;
  #pragma unroll
  for (int i = 0; i < 4; ++i) {
    const int sidx = tid + i * 256;
    float val = sr[sidx] * 0.125f;            // HS^-0.5
    if (causal && sidx > t) val = -1e30f;
    v[i] = val;
    mx = fmaxf(mx, val);
  }
  const float m = block_reduce_max(mx, red);
  float s = 0.0f;
  #pragma unroll
  for (int i = 0; i < 4; ++i) { v[i] = __expf(v[i] - m); s += v[i]; }
  const float inv = 1.0f / block_reduce_sum(s, red);
  __bf16* pr = (__bf16*)scores + row * 2048;  // same bytes as this fp32 row
  #pragma unroll
  for (int i = 0; i < 4; ++i) {
    const int sidx = tid + i * 256;
    pr[sidx] = (__bf16)(v[i] * inv);
  }
}

// ---------------------------------------------------------------------------
// Host-side sequencing
// ---------------------------------------------------------------------------
extern "C" void kernel_launch(void* const* d_in, const int* in_sizes, int n_in,
                              void* d_out, int out_size, void* d_ws, size_t ws_size,
                              hipStream_t stream) {
  const int*   idx  = (const int*)d_in[0];
  const float* tok  = (const float*)d_in[1];
  const float* pos  = (const float*)d_in[2];
  const float* Wq[2] = {(const float*)d_in[3],  (const float*)d_in[11]};
  const float* bq[2] = {(const float*)d_in[4],  (const float*)d_in[12]};
  const float* Wk[2] = {(const float*)d_in[5],  (const float*)d_in[13]};
  const float* bk[2] = {(const float*)d_in[6],  (const float*)d_in[14]};
  const float* Wv[2] = {(const float*)d_in[7],  (const float*)d_in[15]};
  const float* bv[2] = {(const float*)d_in[8],  (const float*)d_in[16]};
  const float* Wp[2] = {(const float*)d_in[9],  (const float*)d_in[17]};
  const float* bp[2] = {(const float*)d_in[10], (const float*)d_in[18]};
  const float* W1    = (const float*)d_in[19];
  const float* b1    = (const float*)d_in[20];
  const float* W2    = (const float*)d_in[21];
  const float* b2    = (const float*)d_in[22];
  const float* lnf_g = (const float*)d_in[23];
  const float* lnf_b = (const float*)d_in[24];
  const float* n1_g  = (const float*)d_in[25];
  const float* n1_b  = (const float*)d_in[26];
  const float* n2_g  = (const float*)d_in[27];
  const float* n2_b  = (const float*)d_in[28];
  const float* nf_g  = (const float*)d_in[29];
  const float* nf_b  = (const float*)d_in[30];
  const float* Wfin  = (const float*)d_in[31];
  const float* bfin  = (const float*)d_in[32];

  // workspace layout (bytes)
  char* ws = (char*)d_ws;
  const long long MT = (long long)B_ * T_;                  // 2048 rows
  float*  x      = (float*)(ws);                            // 8 MB
  __bf16* xn     = (__bf16*)(ws + (8LL << 20));             // 4 MB
  __bf16* qb     = (__bf16*)(ws + (12LL << 20));            // 4 MB [b,h,t,e]
  __bf16* kb     = (__bf16*)(ws + (16LL << 20));            // 4 MB
  __bf16* vb     = (__bf16*)(ws + (20LL << 20));            // 4 MB
  __bf16* ob     = (__bf16*)(ws + (24LL << 20));            // 4 MB [b,t,D]
  __bf16* hb     = (__bf16*)(ws + (28LL << 20));            // 20 MB [2048,5120]
  float*  yb     = (float*)(ws + (48LL << 20));             // 8 MB
  float*  scores = (float*)(ws + (56LL << 20));             // 128 MB [B,H,T,T]
  (void)in_sizes; (void)n_in; (void)out_size; (void)ws_size;

  embed_kernel<<<(B_ * T_ * D_ + 255) / 256, 256, 0, stream>>>(idx, tok, pos, x);

  for (int l = 0; l < L_; ++l) {
    for (int s = 0; s < 2; ++s) {   // s=0: unmasked heads, s=1: causal heads
      // xn = LN(x, n1)
      ln_bf16_kernel<<<(int)MT, 256, 0, stream>>>(x, n1_g + l * D_, n1_b + l * D_, xn);

      // Q, K, V : [2048,1024] = xn @ W[h,D,HS] (+b) -> bf16 [b,h,t,e]
      dim3 gq(D_ / 128, (int)MT / 128, 1);
      gemm_t<BM_HEADED, EP_QKV><<<gq, 256, 0, stream>>>(xn, 0, D_,
          Wq[s] + (long long)l * H_ * D_ * HS_, nullptr, 0, 0,
          bq[s] + l * H_ * HS_, nullptr, nullptr, qb, 0, 0, (int)MT, D_, D_);
      gemm_t<BM_HEADED, EP_QKV><<<gq, 256, 0, stream>>>(xn, 0, D_,
          Wk[s] + (long long)l * H_ * D_ * HS_, nullptr, 0, 0,
          bk[s] + l * H_ * HS_, nullptr, nullptr, kb, 0, 0, (int)MT, D_, D_);
      gemm_t<BM_HEADED, EP_QKV><<<gq, 256, 0, stream>>>(xn, 0, D_,
          Wv[s] + (long long)l * H_ * D_ * HS_, nullptr, 0, 0,
          bv[s] + l * H_ * HS_, nullptr, nullptr, vb, 0, 0, (int)MT, D_, D_);

      // scores[z,t,s'] = q[z] @ k[z]^T   (z = b*H+h), K=HS=64
      dim3 gs(T_ / 128, T_ / 128, B_ * H_);
      gemm_t<BM_BF16T, EP_F32><<<gs, 256, 0, stream>>>(qb, (long long)T_ * HS_, HS_,
          nullptr, kb, (long long)T_ * HS_, HS_,
          nullptr, nullptr, scores, nullptr, (long long)T_ * T_, T_,
          T_, T_, HS_);

      // softmax (scale 1/8, causal iff s==1); writes bf16 probs in place
      softmax_kernel<<<B_ * H_ * T_, 256, 0, stream>>>(scores, s);

      // o[z,t,e] = probs[z] @ v[z]  (A = bf16 probs, lda=2048, K=T)
      dim3 go(1, T_ / 128, B_ * H_);
      gemm_t<BM_BF16, EP_O><<<go, 256, 0, stream>>>((const __bf16*)scores,
          (long long)T_ * T_ * 2, 2048,
          nullptr, vb, (long long)T_ * HS_, HS_,
          nullptr, nullptr, nullptr, ob, 0, HS_,
          T_, HS_, T_);

      // x = x + o @ Wp + bp   (residual fused in epilogue)
      dim3 gp(D_ / 128, (int)MT / 128, 1);
      gemm_t<BM_F32, EP_PROJ><<<gp, 256, 0, stream>>>(ob, 0, D_,
          Wp[s] + (long long)l * D_ * D_, nullptr, 0, D_,
          bp[s] + l * D_, x, x, nullptr, 0, D_, (int)MT, D_, D_);

      // xn = LN(x, n2)
      ln_bf16_kernel<<<(int)MT, 256, 0, stream>>>(x, n2_g + l * D_, n2_b + l * D_, xn);

      // h = leaky_relu(xn @ W1 + b1) -> bf16
      dim3 g1(DFF_ / 128, (int)MT / 128, 1);
      gemm_t<BM_F32, EP_FFN1><<<g1, 256, 0, stream>>>(xn, 0, D_,
          W1 + (long long)l * D_ * DFF_, nullptr, 0, DFF_,
          b1 + l * DFF_, nullptr, nullptr, hb, 0, DFF_, (int)MT, DFF_, D_);

      // y = h @ W2 + b2 -> fp32
      dim3 g2(D_ / 128, (int)MT / 128, 1);
      gemm_t<BM_F32, EP_F32><<<g2, 256, 0, stream>>>(hb, 0, DFF_,
          W2 + (long long)l * DFF_ * D_, nullptr, 0, D_,
          b2 + l * D_, nullptr, yb, nullptr, 0, D_, (int)MT, D_, DFF_);

      // x += LN( LN(y, lnf), n1 (s=0) / n2 (s=1) )
      const float* pg = (s == 0) ? n1_g : n2_g;
      const float* pb = (s == 0) ? n1_b : n2_b;
      ffn_post_kernel<<<(int)MT, 256, 0, stream>>>(yb,
          lnf_g + l * D_, lnf_b + l * D_, pg + l * D_, pb + l * D_, x);
    }
  }

  // final LN + logits
  ln_bf16_kernel<<<(int)MT, 256, 0, stream>>>(x, nf_g, nf_b, xn);
  dim3 gl(V_ / 128, (int)MT / 128, 1);
  gemm_t<BM_F32, EP_F32><<<gl, 256, 0, stream>>>(xn, 0, D_,
      Wfin, nullptr, 0, V_,
      bfin, nullptr, (float*)d_out, nullptr, 0, V_, (int)MT, V_, D_);
}
